// GNN_VGAE_Encoder_58712202936414
// MI455X (gfx1250) — matrix operand order repaired and checked by reading, not compiled, not run on hardware
//
#include <hip/hip_runtime.h>
#include <hip/hip_bf16.h>
#include <cstdint>
#include <cstddef>

// ---------------------------------------------------------------------------
// GAT-VGAE encoder for MI455X (gfx1250, wave32, WMMA).
//
// Attention aggregation dominates (~60 GFLOP of N x N x O GEMM); the N x N
// attention matrix (134MB/layer) is never materialized -> fused flash-style
// streaming softmax + WMMA accumulate. All GEMM operands are laid out so
// every WMMA fragment is a pair of 16-byte loads:
//   A row-major   -> lane's K runs are contiguous (2 x b128)
//   B transposed  -> lane's K run is contiguous   (2 x b128)
// ---------------------------------------------------------------------------

typedef __bf16 bf16;
typedef __attribute__((ext_vector_type(16))) __bf16 v16bf;
typedef __attribute__((ext_vector_type(8)))  __bf16 v8bf;
typedef __attribute__((ext_vector_type(8)))  float  v8f;

// 16-bit fragment maps used in this file (contiguous K, not the ISA 7.12.2 table order):
//  A (16x32): lane<16: M=lane,    K in 0..15   (base 0)
//             lane>=16: M=lane-16, K in 16..31  (base 16)
//  B (32x16): element e -> K = e (+16 for lanes>=16), N = lane & 15
//  C/D (16x16 f32): element r -> row r (lanes<16) / r+8 (lanes>=16), col=lane&15

__device__ __forceinline__ v16bf load_a_frag(const bf16* p /*row start + k0*/,
                                             int lane) {
  const int base = (lane < 16) ? 0 : 16;
  const v8bf lo = *(const v8bf*)(p + base);
  const v8bf hi = *(const v8bf*)(p + base + 8);
  return __builtin_shufflevector(lo, hi, 0, 1, 2, 3, 4, 5, 6, 7,
                                 8, 9, 10, 11, 12, 13, 14, 15);
}
// B from transposed storage: row = output col, 16 contiguous K values.
__device__ __forceinline__ v16bf load_b_frag_t(const bf16* p /*col row + k0*/,
                                               int lane) {
  const int off = (lane < 16) ? 0 : 16;
  return *(const v16bf*)(p + off);
}

// ---------------------------------------------------------------------------
// f32 -> bf16 straight conversion (x once)
// ---------------------------------------------------------------------------
__global__ void cvt_f32_bf16(const float* __restrict__ in, bf16* __restrict__ out,
                             size_t n) {
  size_t i = (size_t)blockIdx.x * blockDim.x + threadIdx.x;
  const size_t stride = (size_t)gridDim.x * blockDim.x;
  for (; i < n; i += stride) out[i] = (bf16)in[i];
}

// ---------------------------------------------------------------------------
// Tiled transpose + f32->bf16: out[b][c][r] = in[b][r][c].  32x32 LDS tiles,
// coalesced loads and stores. blockIdx.y = batch.
// ---------------------------------------------------------------------------
__global__ void transpose_cvt(const float* __restrict__ in, bf16* __restrict__ out,
                              int rows, int cols) {
  __shared__ float t[32][33];
  const int ctiles = cols >> 5;
  const int c0 = (blockIdx.x % ctiles) << 5;
  const int r0 = (blockIdx.x / ctiles) << 5;
  const size_t base = (size_t)blockIdx.y * rows * cols;
  const int tx = threadIdx.x & 31, ty = threadIdx.x >> 5;   // 256 thr: ty 0..7
#pragma unroll
  for (int i = ty; i < 32; i += 8)
    t[i][tx] = in[base + (size_t)(r0 + i) * cols + c0 + tx];
  __syncthreads();
#pragma unroll
  for (int i = ty; i < 32; i += 8)
    out[base + (size_t)(c0 + i) * rows + r0 + tx] = (bf16)t[tx][i];
}

// ---------------------------------------------------------------------------
// hp = A(bf16, MxK row-major) @ W  with W stored transposed (Wt: N x K).
// Each wave computes a 16x64 strip: one A fragment feeds 4 WMMAs.
// ---------------------------------------------------------------------------
__global__ void proj_gemm_bf16(const bf16* __restrict__ A,
                               const bf16* __restrict__ Wt,  // N x K row-major
                               float* __restrict__ C,
                               int M, int K, int N) {
  const int wave = threadIdx.x >> 5;
  const int lane = threadIdx.x & 31;
  const int ct64 = N >> 6;
  const int tile = blockIdx.x * 8 + wave;
  if (tile >= (M >> 4) * ct64) return;        // wave-uniform: EXEC stays full
  const int m0 = (tile / ct64) << 4;
  const int n0 = (tile % ct64) << 6;

  v8f acc[4] = {};
  const int am = m0 + (lane & 15);
  const int bn = n0 + (lane & 15);
  const bf16* arow = A + (size_t)am * K;
  for (int k0 = 0; k0 < K; k0 += 32) {
    const v16bf a = load_a_frag(arow + k0, lane);
#pragma unroll
    for (int t = 0; t < 4; ++t) {
      const v16bf b = load_b_frag_t(Wt + (size_t)(bn + t * 16) * K + k0, lane);
      acc[t] = __builtin_amdgcn_wmma_f32_16x16x32_bf16(false, a, false, b,
                                                       (short)0, acc[t],
                                                       false, false);
    }
  }
#pragma unroll
  for (int t = 0; t < 4; ++t) {
    const int col = bn + t * 16;
#pragma unroll
    for (int r = 0; r < 8; ++r) {
      const int row = m0 + ((lane < 16) ? r : r + 8);
      C[(size_t)row * N + col] = acc[t][r];
    }
  }
}

// ---------------------------------------------------------------------------
// Per-row: s = sum_o tanh(hp)*a_src, d = sum_o tanh(hp)*a_dst.
// One wave per row, shuffle reduction.
// ---------------------------------------------------------------------------
__global__ void row_scores(const float* __restrict__ hp,
                           const float* __restrict__ asrc,
                           const float* __restrict__ adst,
                           float* __restrict__ sout, float* __restrict__ dout,
                           int rows, int O) {
  const int wave = threadIdx.x >> 5;
  const int lane = threadIdx.x & 31;
  const int row = blockIdx.x * 8 + wave;
  if (row >= rows) return;
  float sa = 0.f, da = 0.f;
  for (int c = lane; c < O; c += 32) {
    const float t = tanhf(hp[(size_t)row * O + c]);
    sa += t * asrc[c];
    da += t * adst[c];
  }
#pragma unroll
  for (int k = 16; k >= 1; k >>= 1) {
    sa += __shfl_xor(sa, k, 32);
    da += __shfl_xor(da, k, 32);
  }
  if (lane == 0) { sout[row] = sa; dout[row] = da; }
}

// ---------------------------------------------------------------------------
// Fused masked-softmax aggregation (flash style). Block = 8 waves owns a
// 16-row block; streams 32-wide m tiles. Wave w computes score rows {w,w+8}
// (leaky + adj mask + online softmax, shuffle reductions), P tile goes to LDS
// as bf16; each wave then runs WMMA P @ hp for its output columns, reading
// hp from the transposed bf16 copy so B fragments are contiguous.
// ---------------------------------------------------------------------------
__global__ void gat_aggregate(const bf16* __restrict__ hpT,   // (B, O, N)
                              const float* __restrict__ srow,
                              const float* __restrict__ drow,
                              const unsigned char* __restrict__ adj,
                              const float* __restrict__ bias,
                              bf16* __restrict__ out_bf,      // (B*N, O) or null
                              float* __restrict__ out_f32,    // strided or null
                              int N, int O, int out_stride, int out_col_off,
                              int relu) {
  const int wave = threadIdx.x >> 5;
  const int lane = threadIdx.x & 31;
  const int blocks_per_b = N >> 4;
  const int b  = blockIdx.x / blocks_per_b;
  const int n0 = (blockIdx.x % blocks_per_b) << 4;

  const bf16*  hpb = hpT + (size_t)b * O * N;
  const float* sb  = srow + (size_t)b * N;
  const float* db  = drow + (size_t)b * N;
  const unsigned char* ab = adj + (size_t)b * N * N;

  __shared__ float row_max[16];
  __shared__ float row_sum[16];
  __shared__ float resc[16];
  __shared__ bf16  Pt[16][32];

  if (threadIdx.x < 16) {
    row_max[threadIdx.x] = -1e30f;
    row_sum[threadIdx.x] = 0.f;
    resc[threadIdx.x]    = 1.f;
  }
  __syncthreads();

  const int tpw = O >> 7;                 // col tiles/wave: O=256 -> 2, 128 -> 1
  const int colbase = wave * 16 * tpw;
  v8f acc[2] = {};

  const int rA = n0 + wave, rB = n0 + wave + 8;
  const float sA = sb[rA], sB = sb[rB];
  const unsigned char* adjA = ab + (size_t)rA * N;
  const unsigned char* adjB = ab + (size_t)rB * N;

  for (int m0 = 0; m0 < N; m0 += 32) {
    // ---- phase 1: scores + online-softmax bookkeeping ----
    if (m0 + 32 < N) {                     // prefetch next adjacency tile
      __builtin_prefetch(adjA + m0 + 32 + lane, 0, 0);
      __builtin_prefetch(adjB + m0 + 32 + lane, 0, 0);
    }
    const float dv = db[m0 + lane];
    const unsigned char aA = adjA[m0 + lane];
    const unsigned char aB = adjB[m0 + lane];
    float e1 = sA + dv; e1 = (e1 >= 0.f) ? e1 : 0.2f * e1; e1 = aA ? e1 : -1e30f;
    float e2 = sB + dv; e2 = (e2 >= 0.f) ? e2 : 0.2f * e2; e2 = aB ? e2 : -1e30f;
    float m1 = e1, m2 = e2;
#pragma unroll
    for (int k = 16; k >= 1; k >>= 1) {
      m1 = fmaxf(m1, __shfl_xor(m1, k, 32));
      m2 = fmaxf(m2, __shfl_xor(m2, k, 32));
    }
    const float Mo1 = row_max[wave], Mo2 = row_max[wave + 8];
    const float Mn1 = fmaxf(Mo1, m1), Mn2 = fmaxf(Mo2, m2);
    const float p1 = aA ? __expf(e1 - Mn1) : 0.f;   // masked lanes contribute 0
    const float p2 = aB ? __expf(e2 - Mn2) : 0.f;
    Pt[wave][lane]     = (bf16)p1;
    Pt[wave + 8][lane] = (bf16)p2;
    float t1 = p1, t2 = p2;
#pragma unroll
    for (int k = 16; k >= 1; k >>= 1) {
      t1 += __shfl_xor(t1, k, 32);
      t2 += __shfl_xor(t2, k, 32);
    }
    if (lane == 0) {
      const float f1 = __expf(Mo1 - Mn1);
      const float f2 = __expf(Mo2 - Mn2);
      row_sum[wave]     = row_sum[wave]     * f1 + t1;
      row_sum[wave + 8] = row_sum[wave + 8] * f2 + t2;
      row_max[wave] = Mn1; row_max[wave + 8] = Mn2;
      resc[wave] = f1;     resc[wave + 8] = f2;
    }
    __syncthreads();

    // ---- phase 2: accumulate P @ hp tile with WMMA ----
    float rs[8];
#pragma unroll
    for (int r = 0; r < 8; ++r) rs[r] = resc[(lane < 16) ? r : r + 8];

    const v16bf afr = load_a_frag(&Pt[lane & 15][0], lane);  // 2 x ds b128

    for (int t = 0; t < tpw; ++t) {
#pragma unroll
      for (int r = 0; r < 8; ++r) acc[t][r] *= rs[r];
      const int col = colbase + t * 16 + (lane & 15);
      const v16bf bfr = load_b_frag_t(hpb + (size_t)col * N + m0, lane);
      acc[t] = __builtin_amdgcn_wmma_f32_16x16x32_bf16(false, afr, false, bfr,
                                                       (short)0, acc[t],
                                                       false, false);
    }
    __syncthreads();   // protect Pt/resc before next iteration's writes
  }

  // ---- epilogue: normalize, bias, (relu), store ----
  float Ls[8];
#pragma unroll
  for (int r = 0; r < 8; ++r) Ls[r] = row_sum[(lane < 16) ? r : r + 8];
  for (int t = 0; t < tpw; ++t) {
    const int col = colbase + t * 16 + (lane & 15);
    const float bv = bias[col];
#pragma unroll
    for (int r = 0; r < 8; ++r) {
      const int row = n0 + ((lane < 16) ? r : r + 8);
      float v = acc[t][r] / Ls[r] + bv;
      if (relu) v = fmaxf(v, 0.f);
      if (out_bf)  out_bf[(size_t)(b * N + row) * O + col] = (bf16)v;
      if (out_f32) out_f32[(size_t)(b * N + row) * out_stride + out_col_off + col] = v;
    }
  }
}

// ---------------------------------------------------------------------------
// Host orchestration
// ---------------------------------------------------------------------------
namespace {
constexpr int Bz = 8, Nn = 2048, BN = Bz * Nn;

struct Ws {
  bf16* bufA; bf16* bufB; bf16* wT;
  float* hp32; bf16* hpT; float* sbuf; float* dbuf;
};

static Ws carve(void* d_ws) {
  char* p = (char*)d_ws;
  Ws w;
  w.bufA = (bf16*)p;  p += (size_t)BN * 512 * sizeof(bf16);
  w.bufB = (bf16*)p;  p += (size_t)BN * 256 * sizeof(bf16);
  w.wT   = (bf16*)p;  p += (size_t)512 * 256 * sizeof(bf16);
  w.hp32 = (float*)p; p += (size_t)BN * 256 * sizeof(float);
  w.hpT  = (bf16*)p;  p += (size_t)BN * 256 * sizeof(bf16);
  w.sbuf = (float*)p; p += (size_t)BN * sizeof(float);
  w.dbuf = (float*)p; p += (size_t)BN * sizeof(float);
  return w;                                   // ~51 MB total
}

static void run_layer(const bf16* in16, int Fin, int O,
                      const float* W, const float* asrc, const float* adst,
                      const float* bias, const unsigned char* adj,
                      bf16* out_bf, float* out_f32, int out_stride, int out_col,
                      int relu, const Ws& ws, hipStream_t stream) {
  // W (Fin x O) -> Wt (O x Fin) bf16
  {
    dim3 g((Fin / 32) * (O / 32), 1);
    transpose_cvt<<<g, 256, 0, stream>>>(W, ws.wT, Fin, O);
  }
  // hp = in @ W  (f32 out)
  const int blocks = (BN / 16) * (O / 64) / 8;
  proj_gemm_bf16<<<blocks, 256, 0, stream>>>(in16, ws.wT, ws.hp32, BN, Fin, O);
  // per-row scores
  row_scores<<<BN / 8, 256, 0, stream>>>(ws.hp32, asrc, adst,
                                         ws.sbuf, ws.dbuf, BN, O);
  // hp (B,N,O) -> hpT (B,O,N) bf16
  {
    dim3 g((O / 32) * (Nn / 32), Bz);
    transpose_cvt<<<g, 256, 0, stream>>>(ws.hp32, ws.hpT, Nn, O);
  }
  // fused masked softmax + aggregation
  gat_aggregate<<<Bz * (Nn / 16), 256, 0, stream>>>(
      ws.hpT, ws.sbuf, ws.dbuf, adj, bias, out_bf, out_f32,
      Nn, O, out_stride, out_col, relu);
}
}  // namespace

extern "C" void kernel_launch(void* const* d_in, const int* in_sizes, int n_in,
                              void* d_out, int out_size, void* d_ws, size_t ws_size,
                              hipStream_t stream) {
  const float* x = (const float*)d_in[0];
  const unsigned char* adj = (const unsigned char*)d_in[1];
  const float* w0[4] = {(const float*)d_in[2],  (const float*)d_in[3],
                        (const float*)d_in[4],  (const float*)d_in[5]};
  const float* w1[4] = {(const float*)d_in[6],  (const float*)d_in[7],
                        (const float*)d_in[8],  (const float*)d_in[9]};
  const float* wd[4] = {(const float*)d_in[10], (const float*)d_in[11],
                        (const float*)d_in[12], (const float*)d_in[13]};
  const float* wy[4] = {(const float*)d_in[14], (const float*)d_in[15],
                        (const float*)d_in[16], (const float*)d_in[17]};
  const float* wm[4] = {(const float*)d_in[18], (const float*)d_in[19],
                        (const float*)d_in[20], (const float*)d_in[21]};
  float* out = (float*)d_out;
  Ws ws = carve(d_ws);

  const size_t xn = (size_t)BN * 512;
  cvt_f32_bf16<<<(int)((xn + 255) / 256), 256, 0, stream>>>(x, ws.bufA, xn);

  run_layer(ws.bufA, 512, 256, w0[0], w0[1], w0[2], w0[3], adj,
            ws.bufB, nullptr, 0, 0, 1, ws, stream);
  run_layer(ws.bufB, 256, 256, w1[0], w1[1], w1[2], w1[3], adj,
            ws.bufA, nullptr, 0, 0, 1, ws, stream);
  run_layer(ws.bufA, 256, 128, wd[0], wd[1], wd[2], wd[3], adj,
            nullptr, out, 384, 0, 0, ws, stream);
  run_layer(ws.bufA, 256, 128, wy[0], wy[1], wy[2], wy[3], adj,
            nullptr, out, 384, 128, 0, ws, stream);
  run_layer(ws.bufA, 256, 128, wm[0], wm[1], wm[2], wm[3], adj,
            nullptr, out, 384, 256, 0, ws, stream);
}
